// VanillaVQ_24292335026189
// MI455X (gfx1250) — matrix operand order repaired and checked by reading, not compiled, not run on hardware
//
#include <hip/hip_runtime.h>

#define DIM   512
#define NROWS 32768
#define KCODE 8192
#define CHUNK 32                  // codebook rows per LDS tile (32KB), double-buffered
#define NCHUNK (KCODE / CHUNK)
#define ROWS_PER_WG 128           // 8 waves x 16 rows
#define UNITS_PER_ROW (DIM / 8)   // 16-byte units per codebook row (64)

typedef __attribute__((ext_vector_type(16))) _Float16 v16h;
typedef __attribute__((ext_vector_type(8)))  _Float16 v8h;
typedef __attribute__((ext_vector_type(4)))  _Float16 v4h;
typedef __attribute__((ext_vector_type(8)))  float    v8f;
typedef __attribute__((ext_vector_type(4)))  float    v4f;
typedef __attribute__((ext_vector_type(4)))  int      v4i;

#if defined(__gfx1250__) && __has_builtin(__builtin_amdgcn_global_load_async_to_lds_b128)
#define HAVE_ASYNC_LDS 1
typedef __attribute__((address_space(1))) v4i* gv4i_p;   // global
typedef __attribute__((address_space(3))) v4i* lv4i_p;   // LDS
#else
#define HAVE_ASYNC_LDS 0
#endif

// ---------------- Kernel 1: z (f32) -> z (f16) ----------------
__global__ __launch_bounds__(256) void vq_cvt_z(const float* __restrict__ z,
                                                _Float16* __restrict__ zh) {
    const int i = (blockIdx.x * 256 + threadIdx.x) * 8;
    v4f a = *(const v4f*)(z + i);
    v4f b = *(const v4f*)(z + i + 4);
    v8h o;
    o[0] = (_Float16)a[0]; o[1] = (_Float16)a[1];
    o[2] = (_Float16)a[2]; o[3] = (_Float16)a[3];
    o[4] = (_Float16)b[0]; o[5] = (_Float16)b[1];
    o[6] = (_Float16)b[2]; o[7] = (_Float16)b[3];
    *(v8h*)(zh + i) = o;
}

// ---- Kernel 2: codebook -> scaled f16 (x8192) and -4096*||c||^2 ----
__global__ __launch_bounds__(256) void vq_prep_cb(const float* __restrict__ cb,
                                                  _Float16* __restrict__ cbh,
                                                  float* __restrict__ csq) {
    const int code = (blockIdx.x * 256 + threadIdx.x) >> 5;   // one wave per code
    const int lane = threadIdx.x & 31;
    const float* row = cb + (size_t)code * DIM;
    _Float16* orow = cbh + (size_t)code * DIM;
    float ss = 0.f;
#pragma unroll
    for (int i = 0; i < 4; ++i) {
        const int col = (i * 32 + lane) * 4;
        v4f v = *(const v4f*)(row + col);
        ss += v[0]*v[0] + v[1]*v[1] + v[2]*v[2] + v[3]*v[3];
        v4h h;
        h[0] = (_Float16)(v[0] * 8192.f); h[1] = (_Float16)(v[1] * 8192.f);
        h[2] = (_Float16)(v[2] * 8192.f); h[3] = (_Float16)(v[3] * 8192.f);
        *(v4h*)(orow + col) = h;
    }
#pragma unroll
    for (int off = 16; off > 0; off >>= 1)
        ss += __shfl_down(ss, off, 32);
    if (lane == 0) csq[code] = -4096.f * ss;
}

// ---- chunk staging: global f16 codebook -> XOR-swizzled LDS tile ----
__device__ __forceinline__ void stage_chunk(const _Float16* __restrict__ cbh,
                                            _Float16* __restrict__ buf,
                                            int cBase, int tid) {
#if HAVE_ASYNC_LDS
#pragma unroll
    for (int i = 0; i < (CHUNK * UNITS_PER_ROW) / 256; ++i) {      // 8 iters
        const int u    = tid + i * 256;
        const int row  = u >> 6;
        const int colU = u & 63;
        __builtin_amdgcn_global_load_async_to_lds_b128(
            (gv4i_p)(cbh + (size_t)(cBase + row) * DIM + colU * 8),
            (lv4i_p)(buf + row * DIM + ((colU ^ (row & 15)) * 8)),
            0, 0);
    }
#else
    // batched fallback: 8 loads in flight before the stores
    v8h t[8];
#pragma unroll
    for (int j = 0; j < 8; ++j) {
        const int u = tid + j * 256;
        const int row = u >> 6, colU = u & 63;
        t[j] = *(const v8h*)(cbh + (size_t)(cBase + row) * DIM + colU * 8);
    }
#pragma unroll
    for (int j = 0; j < 8; ++j) {
        const int u = tid + j * 256;
        const int row = u >> 6, colU = u & 63;
        *(v8h*)(buf + row * DIM + ((colU ^ (row & 15)) * 8)) = t[j];
    }
#endif
}

__device__ __forceinline__ void wait_async_copies() {
#if HAVE_ASYNC_LDS
#if __has_builtin(__builtin_amdgcn_s_wait_asynccnt)
    __builtin_amdgcn_s_wait_asynccnt(0);
#else
    asm volatile("s_wait_asynccnt 0x0" ::: "memory");
#endif
#endif
}

// ---------------- Kernel 3: fused cross-GEMM + argmin + gather ----------------
// acc = z_scaled_cross - 4096*||c||^2  ==>  argmin(d2) == argmax(acc)
__global__ __launch_bounds__(256) void vq_main(const _Float16* __restrict__ zh,
                                               const _Float16* __restrict__ cbh,
                                               const float* __restrict__ csq,
                                               const float* __restrict__ cb,
                                               float* __restrict__ zq,
                                               float* __restrict__ idxf) {
    extern __shared__ _Float16 lds[];   // 2 * CHUNK*DIM halfs (double buffer)

    const int tid  = threadIdx.x;
    const int lane = tid & 31;
    const int m    = lane & 15;         // row-in-tile (A: code, B/C: z-row)
    const int hb   = lane >> 4;         // lane half
    const int wave = tid >> 5;
    const int rowBase = blockIdx.x * ROWS_PER_WG + wave * 16;

    // Resident B fragments: this wave's 16 z-rows, full depth 512.
    v16h bfrag[16];
    {
        const _Float16* zr = zh + (size_t)(rowBase + m) * DIM;
#pragma unroll
        for (int kk = 0; kk < 16; ++kk) {
            const _Float16* p = zr + kk * 32 + hb * 8;
            v8h lo = *(const v8h*)(p);
            v8h hi = *(const v8h*)(p + 16);
            bfrag[kk] = __builtin_shufflevector(lo, hi,
                0,1,2,3,4,5,6,7,8,9,10,11,12,13,14,15);
        }
    }

    float bestVal = -3.0e38f;
    int   bestIdx = 0;

    _Float16* buf0 = lds;
    _Float16* buf1 = lds + CHUNK * DIM;

    stage_chunk(cbh, buf0, 0, tid);
    wait_async_copies();
    __syncthreads();

    for (int c = 0; c < NCHUNK; ++c) {
        _Float16* cur = (c & 1) ? buf1 : buf0;
        if (c + 1 < NCHUNK)
            stage_chunk(cbh, (c & 1) ? buf0 : buf1, (c + 1) * CHUNK, tid);

        const int cBase = c * CHUNK;
#pragma unroll
        for (int jt = 0; jt < CHUNK / 16; ++jt) {
            // accumulator pre-loaded with -4096*||c||^2 (C: VGPR r -> M = hb*8+r)
            const float* qp = csq + cBase + jt * 16 + hb * 8;
            v4f q0 = *(const v4f*)(qp);
            v4f q1 = *(const v4f*)(qp + 4);
            v8f acc = { q0[0], q0[1], q0[2], q0[3], q1[0], q1[1], q1[2], q1[3] };

            const _Float16* arow = cur + (jt * 16 + m) * DIM;

            // software-pipelined A-fragment loads, prefetch distance 2
            v8h alo[3], ahi[3];
#define LOADA(KK, S) do {                                     \
                const int c1_ = ((((KK) * 4 + hb)     ^ m) * 8); \
                const int c2_ = ((((KK) * 4 + hb + 2) ^ m) * 8); \
                alo[S] = *(const v8h*)(arow + c1_);              \
                ahi[S] = *(const v8h*)(arow + c2_);              \
            } while (0)
            LOADA(0, 0);
            LOADA(1, 1);
#pragma unroll
            for (int kk = 0; kk < 16; ++kk) {
                if (kk + 2 < 16) LOADA(kk + 2, (kk + 2) % 3);
                v16h afrag = __builtin_shufflevector(alo[kk % 3], ahi[kk % 3],
                    0,1,2,3,4,5,6,7,8,9,10,11,12,13,14,15);
                acc = __builtin_amdgcn_wmma_f32_16x16x32_f16(
                          false, afrag, false, bfrag[kk],
                          (short)0, acc, false, false);
            }
#undef LOADA

#pragma unroll
            for (int r = 0; r < 8; ++r) {
                const int code = cBase + jt * 16 + hb * 8 + r;
                if (acc[r] > bestVal) { bestVal = acc[r]; bestIdx = code; }
            }
        }
        wait_async_copies();
        __syncthreads();
    }

    // Merge lane L with L+16 (codes 0-7 vs 8-15 of each tile); smaller index wins ties.
    {
        const float oV = __shfl_xor(bestVal, 16, 32);
        const int   oI = __shfl_xor(bestIdx, 16, 32);
        if (oV > bestVal || (oV == bestVal && oI < bestIdx)) { bestVal = oV; bestIdx = oI; }
    }

    if (lane < 16) idxf[rowBase + m] = (float)bestIdx;

    // Gather z_q = codebook[bestIdx] (original fp32), cooperative per row.
    for (int r = 0; r < 16; ++r) {
        const int idx = __shfl(bestIdx, r, 32);
        const v4f* src = (const v4f*)(cb + (size_t)idx * DIM);
        v4f* dst = (v4f*)(zq + (size_t)(rowBase + r) * DIM);
#pragma unroll
        for (int i = 0; i < 4; ++i)
            dst[lane + i * 32] = src[lane + i * 32];
    }
}

extern "C" void kernel_launch(void* const* d_in, const int* in_sizes, int n_in,
                              void* d_out, int out_size, void* d_ws, size_t ws_size,
                              hipStream_t stream) {
    const float* z  = (const float*)d_in[0];
    const float* cb = (const float*)d_in[1];

    // workspace layout: [z_f16: 32MB][cb_f16: 8MB][csq: 32KB]
    _Float16* zh  = (_Float16*)d_ws;
    _Float16* cbh = (_Float16*)((char*)d_ws + (size_t)NROWS * DIM * 2);
    float*    csq = (float*)((char*)d_ws + (size_t)NROWS * DIM * 2 + (size_t)KCODE * DIM * 2);

    float* zq   = (float*)d_out;
    float* idxf = zq + (size_t)NROWS * DIM;

    vq_cvt_z <<<(NROWS * DIM / 8) / 256, 256, 0, stream>>>(z, zh);
    vq_prep_cb<<<(KCODE * 32) / 256,     256, 0, stream>>>(cb, cbh, csq);
    vq_main  <<<NROWS / ROWS_PER_WG,     256,
               2 * CHUNK * DIM * sizeof(_Float16), stream>>>(zh, cbh, csq, cb, zq, idxf);
}
